// DynamicDepthTransformer_65687229825598
// MI455X (gfx1250) — matrix operand order repaired and checked by reading, not compiled, not run on hardware
//
#include <hip/hip_runtime.h>
#include <stdint.h>

#define DEVFN __device__ __forceinline__

// CDNA5 async global->LDS staging (guarded: falls back to manual staging if
// the toolchain doesn't declare the gfx1250 async builtins).
// Probe result: the builtin takes pointers to v4i32 (vector_size(16) int).
#if defined(__AMDGCN__) && __has_builtin(__builtin_amdgcn_global_load_async_to_lds_b128) && __has_builtin(__builtin_amdgcn_s_wait_asynccnt)
#define USE_ASYNC_LDS 1
typedef int v4i_vs __attribute__((vector_size(16)));
typedef __attribute__((address_space(1))) v4i_vs* g_b128_ptr;   // global src
typedef __attribute__((address_space(3))) v4i_vs* l_b128_ptr;   // LDS dst
#else
#define USE_ASYNC_LDS 0
#endif

constexpr int B = 4, T = 256, D = 1024, H = 16, HD = 64, F = 4096, L = 6, V = 32000;
constexpr float EXIT_THR = 0.9f, SKIP_THR = 0.1f, TEMP = 1.0f, EPS = 1e-6f;
constexpr int MIN_LAYERS = 1;
constexpr int MR = B * T;            // 1024 rows of activations

typedef __attribute__((ext_vector_type(16))) __bf16 v16bf;
typedef __attribute__((ext_vector_type(8)))  float  v8f;

DEVFN uint16_t f2bf(float f) {
  union { float f; uint32_t u; } c{f};
  uint32_t u = c.u;
  u += 0x7FFFu + ((u >> 16) & 1u);      // round-to-nearest-even
  return (uint16_t)(u >> 16);
}
DEVFN float bf2f(uint16_t h) {
  union { uint32_t u; float f; } c{(uint32_t)h << 16};
  return c.f;
}

// ---------------------------------------------------------------------------
// Generic activation(bf16) x weight(fp32, converted on-the-fly) WMMA GEMM.
// Block tile 64(M) x 256(N) x 32(K); 8 waves, each wave owns a 32x64 tile
// -> 8x v_wmma_f32_16x16x32_bf16 per K-step.
// A tile staged via GLOBAL_LOAD_ASYNC_TO_LDS_B128 when available.
// ---------------------------------------------------------------------------
enum { EP_F32 = 0, EP_BF16 = 1, EP_RESID = 2, EP_VT = 3 };

template<int EP>
__global__ __launch_bounds__(256)
void gemm_bf16(const uint16_t* __restrict__ A, const float* __restrict__ W,
               int M, int N, int K,
               float* __restrict__ outF, uint16_t* __restrict__ outB,
               const float* __restrict__ resid,
               const int* __restrict__ gate, const int* __restrict__ batch_pred)
{
  if (gate && gate[0]) return;                     // layer skipped (device flag)
  constexpr int BM = 64, BN = 256, BK = 32;
  const int m0 = blockIdx.y * BM;
  const int n0 = blockIdx.x * BN;
  if (batch_pred && batch_pred[m0 / T] == 0) return;   // batch not selected

  __shared__ __align__(16) uint16_t sA[BM * BK];       // [m][k] bf16      (4 KB)
  __shared__ __align__(16) uint16_t sB[BN * BK];       // transposed [n][k] (16 KB)

  const int tid  = threadIdx.x;
  const int lane = tid & 31;
  const int wid  = tid >> 5;
  const int mr = (wid & 1) * 32;       // wave M offset in block tile
  const int nc = (wid >> 1) * 64;      // wave N offset in block tile
  const int half = lane >> 4, nl = lane & 15;

  v8f acc[2][4];
  #pragma unroll
  for (int i = 0; i < 2; ++i)
    #pragma unroll
    for (int j = 0; j < 4; ++j)
      #pragma unroll
      for (int r = 0; r < 8; ++r) acc[i][j][r] = 0.0f;

#if !USE_ASYNC_LDS
  uint32_t* sA32 = (uint32_t*)sA;
#endif

  for (int k0 = 0; k0 < K; k0 += BK) {
#if USE_ASYNC_LDS
    {
      // one 16-byte async transfer per thread covers the 4 KB A tile
      const int row = tid >> 2, c4 = tid & 3;
      const uint16_t* gp = A + (size_t)(m0 + row) * K + k0 + c4 * 8;
      uint16_t* lp = sA + row * BK + c4 * 8;
      __builtin_amdgcn_global_load_async_to_lds_b128(
          (g_b128_ptr)(void*)gp, (l_b128_ptr)(void*)lp, 0, 0);
    }
#else
    #pragma unroll
    for (int u = 0; u < 4; ++u) {
      int q = tid + u * 256;
      int row = q >> 4, c2 = q & 15;
      sA32[row * 16 + c2] =
          *(const uint32_t*)(A + (size_t)(m0 + row) * K + k0 + c2 * 2);
    }
#endif
    // stage W tile fp32 -> bf16, transposed [n][k]
    #pragma unroll
    for (int u = 0; u < 32; ++u) {
      int f = tid + u * 256;
      int kk = f >> 8, nn = f & 255;
      float w = W[(size_t)(k0 + kk) * N + n0 + nn];
      sB[nn * BK + kk] = f2bf(w);
    }
    if (k0 + BK < K) {   // emits global_prefetch_b8 for next weight tile
      __builtin_prefetch(W + (size_t)(k0 + BK) * N + n0 + tid, 0, 0);
    }
#if USE_ASYNC_LDS
    __builtin_amdgcn_s_wait_asynccnt(0);
#endif
    __syncthreads();

    v16bf a0, a1, bf[4];
    {
      const uint32_t* p0 = (const uint32_t*)(sA + (mr + nl) * BK);
      const uint32_t* p1 = (const uint32_t*)(sA + (mr + 16 + nl) * BK);
      uint32_t* a0p = (uint32_t*)&a0; uint32_t* a1p = (uint32_t*)&a1;
      const int ab = half * 8;
      #pragma unroll
      for (int j = 0; j < 4; ++j) {
        a0p[j] = p0[ab + j];  a0p[4 + j] = p0[4 + ab + j];
        a1p[j] = p1[ab + j];  a1p[4 + j] = p1[4 + ab + j];
      }
      const int kb = half * 8;
      #pragma unroll
      for (int j = 0; j < 4; ++j) {
        const uint32_t* q0 = (const uint32_t*)(sB + (nc + j * 16 + nl) * BK);
        uint32_t* bp = (uint32_t*)&bf[j];
        #pragma unroll
        for (int u = 0; u < 8; ++u) bp[u] = q0[kb + u];
      }
    }
    #pragma unroll
    for (int j = 0; j < 4; ++j) {
      acc[0][j] = __builtin_amdgcn_wmma_f32_16x16x32_bf16(false, a0, false, bf[j], (short)0, acc[0][j], false, false);
      acc[1][j] = __builtin_amdgcn_wmma_f32_16x16x32_bf16(false, a1, false, bf[j], (short)0, acc[1][j], false, false);
    }
    __syncthreads();
  }

  #pragma unroll
  for (int i = 0; i < 2; ++i)
    #pragma unroll
    for (int j = 0; j < 4; ++j)
      #pragma unroll
      for (int r = 0; r < 8; ++r) {
        int m = m0 + mr + i * 16 + r + half * 8;
        int n = n0 + nc + j * 16 + nl;
        float v = acc[i][j][r];
        if (EP == EP_F32)       outF[(size_t)m * N + n] = v;
        else if (EP == EP_BF16) outB[(size_t)m * N + n] = f2bf(v);
        else if (EP == EP_RESID) outF[(size_t)m * N + n] = resid[(size_t)m * N + n] + v;
        else { // EP_VT: store V projection as [b][h][hd][t] bf16 for PV WMMA
          int b_ = m >> 8, t_ = m & 255;
          int h_ = n >> 6, hd_ = n & 63;
          outB[(((size_t)b_ * H + h_) * HD + hd_) * T + t_] = f2bf(v);
        }
      }
}

// ---------------------------------------------------------------------------
// Attention: scores = q k^T / sqrt(HD) with causal mask (one wave per 16x16)
// ---------------------------------------------------------------------------
__global__ __launch_bounds__(256)
void attn_scores(const uint16_t* __restrict__ qb, const uint16_t* __restrict__ kb,
                 float* __restrict__ scores, const int* __restrict__ gate)
{
  if (gate && gate[0]) return;
  const int gw   = (blockIdx.x * 256 + threadIdx.x) >> 5;
  const int lane = threadIdx.x & 31;
  constexpr int QT = T / 16, KT = T / 16;
  const int bh = gw / (QT * KT);
  const int rem = gw % (QT * KT);
  const int qt = rem / KT, kt = rem % KT;
  const int b = bh / H, h = bh % H;
  const int half = lane >> 4, nl = lane & 15;
  float* srow = scores + (size_t)bh * T * T;

  if (kt > qt) {                                   // fully masked tile
    #pragma unroll
    for (int r = 0; r < 8; ++r)
      srow[(size_t)(qt * 16 + r + half * 8) * T + kt * 16 + nl] = -1e30f;
    return;
  }
  v8f acc;
  #pragma unroll
  for (int r = 0; r < 8; ++r) acc[r] = 0.0f;
  #pragma unroll
  for (int kc = 0; kc < HD; kc += 32) {
    v16bf a, bm;
    uint32_t* ap = (uint32_t*)&a; uint32_t* bp = (uint32_t*)&bm;
    const uint32_t* p  = (const uint32_t*)(qb + (size_t)(b * T + qt * 16 + nl) * D + h * HD + kc);
    const uint32_t* pk = (const uint32_t*)(kb + (size_t)(b * T + kt * 16 + nl) * D + h * HD + kc);
    const int ab = half * 8, kb8 = half * 8;
    #pragma unroll
    for (int j = 0; j < 4; ++j) { ap[j] = p[ab + j]; ap[4 + j] = p[4 + ab + j]; }
    #pragma unroll
    for (int j = 0; j < 8; ++j) bp[j] = pk[kb8 + j];
    acc = __builtin_amdgcn_wmma_f32_16x16x32_bf16(false, a, false, bm, (short)0, acc, false, false);
  }
  #pragma unroll
  for (int r = 0; r < 8; ++r) {
    int qr = qt * 16 + r + half * 8;
    int kc = kt * 16 + nl;
    float v = acc[r] * 0.125f;                     // 1/sqrt(64)
    if (kc > qr) v = -1e30f;
    srow[(size_t)qr * T + kc] = v;
  }
}

__global__ __launch_bounds__(128)
void softmax_rows(const float* __restrict__ scores, uint16_t* __restrict__ attb,
                  const int* __restrict__ gate)
{
  if (gate && gate[0]) return;
  const int row = blockIdx.x;
  const float* s = scores + (size_t)row * T;
  uint16_t* o = attb + (size_t)row * T;
  const int tid = threadIdx.x;
  float v0 = s[tid], v1 = s[tid + 128];
  __shared__ float red[128];
  red[tid] = fmaxf(v0, v1); __syncthreads();
  for (int st = 64; st > 0; st >>= 1) { if (tid < st) red[tid] = fmaxf(red[tid], red[tid + st]); __syncthreads(); }
  float mx = red[0]; __syncthreads();
  float e0 = __expf(v0 - mx), e1 = __expf(v1 - mx);
  red[tid] = e0 + e1; __syncthreads();
  for (int st = 64; st > 0; st >>= 1) { if (tid < st) red[tid] += red[tid + st]; __syncthreads(); }
  float inv = 1.0f / red[0];
  o[tid] = f2bf(e0 * inv);
  o[tid + 128] = f2bf(e1 * inv);
}

// att @ v, one wave per (16 queries x 16 head-dims), K-loop over key positions
__global__ __launch_bounds__(256)
void attn_pv(const uint16_t* __restrict__ attb, const uint16_t* __restrict__ vT,
             uint16_t* __restrict__ obf, const int* __restrict__ gate)
{
  if (gate && gate[0]) return;
  const int gw   = (blockIdx.x * 256 + threadIdx.x) >> 5;
  const int lane = threadIdx.x & 31;
  constexpr int QT = T / 16, NT = HD / 16;
  const int bh = gw / (QT * NT);
  const int rem = gw % (QT * NT);
  const int qt = rem / NT, nt = rem % NT;
  const int b = bh / H, h = bh % H;
  const int half = lane >> 4, nl = lane & 15;
  v8f acc;
  #pragma unroll
  for (int r = 0; r < 8; ++r) acc[r] = 0.0f;
  const uint16_t* arow = attb + (size_t)bh * T * T;
  const int kcmax = ((qt + 1) * 16 + 31) / 32;     // causal: weights beyond q are 0
  for (int kc = 0; kc < kcmax; ++kc) {
    v16bf a, bm;
    uint32_t* ap = (uint32_t*)&a; uint32_t* bp = (uint32_t*)&bm;
    const uint32_t* p  = (const uint32_t*)(arow + (size_t)(qt * 16 + nl) * T + kc * 32);
    const uint32_t* pv = (const uint32_t*)(vT + (((size_t)(b * H + h) * HD) + nt * 16 + nl) * T + kc * 32);
    const int ab = half * 8, kb8 = half * 8;
    #pragma unroll
    for (int j = 0; j < 4; ++j) { ap[j] = p[ab + j]; ap[4 + j] = p[4 + ab + j]; }
    #pragma unroll
    for (int j = 0; j < 8; ++j) bp[j] = pv[kb8 + j];
    acc = __builtin_amdgcn_wmma_f32_16x16x32_bf16(false, a, false, bm, (short)0, acc, false, false);
  }
  #pragma unroll
  for (int r = 0; r < 8; ++r) {
    int q = qt * 16 + r + half * 8;
    obf[(size_t)(b * T + q) * D + h * HD + nt * 16 + nl] = f2bf(acc[r]);
  }
}

// ---------------------------------------------------------------------------
// Elementwise / norm / gating kernels
// ---------------------------------------------------------------------------
__global__ void embed_kernel(const int* __restrict__ ids, const float* __restrict__ emb,
                             float* __restrict__ x)
{
  const int row = blockIdx.x;
  const int id = ids[row];
  const float4* src = (const float4*)(emb + (size_t)id * D);
  float4* dst = (float4*)(x + (size_t)row * D);
  dst[threadIdx.x] = src[threadIdx.x];
}

__global__ __launch_bounds__(256)
void rmsnorm_kernel(const float* __restrict__ x, const float* __restrict__ w,
                    uint16_t* __restrict__ outB, const int* __restrict__ gate)
{
  if (gate && gate[0]) return;
  const int row = blockIdx.x, tid = threadIdx.x;
  const float* xr = x + (size_t)row * D;
  float vals[4], ss = 0.0f;
  #pragma unroll
  for (int u = 0; u < 4; ++u) { vals[u] = xr[tid + u * 256]; ss += vals[u] * vals[u]; }
  __shared__ float red[256];
  red[tid] = ss; __syncthreads();
  for (int st = 128; st > 0; st >>= 1) { if (tid < st) red[tid] += red[tid + st]; __syncthreads(); }
  float sc = rsqrtf(red[0] / (float)D + EPS);
  #pragma unroll
  for (int u = 0; u < 4; ++u) {
    int c = tid + u * 256;
    outB[(size_t)row * D + c] = f2bf(vals[u] * w[c] * sc);
  }
}

__global__ void rope_kernel(const float* __restrict__ inF, uint16_t* __restrict__ outB,
                            const int* __restrict__ gate)
{
  if (gate && gate[0]) return;
  const int idx = blockIdx.x * blockDim.x + threadIdx.x;  // B*T*H*32
  const int i = idx & 31;
  const int h = (idx >> 5) & (H - 1);
  const int bt = idx >> 9;
  const int t = bt & (T - 1);
  const float inv = __expf(-(float)(2 * i) * (9.210340371976184f / 64.0f)); // 10000^(-2i/64)
  const float f = (float)t * inv;
  float s, c;
  __sincosf(f, &s, &c);
  const size_t base = (size_t)bt * D + h * HD;
  float x1 = inF[base + i], x2 = inF[base + 32 + i];
  outB[base + i]      = f2bf(x1 * c - x2 * s);
  outB[base + 32 + i] = f2bf(x1 * s + x2 * c);
}

__global__ void silu_mul_kernel(const float* __restrict__ g1, const float* __restrict__ g3,
                                uint16_t* __restrict__ out, const int* __restrict__ gate, int n)
{
  if (gate && gate[0]) return;
  const int i = blockIdx.x * blockDim.x + threadIdx.x;
  if (i >= n) return;
  float a = g1[i];
  out[i] = f2bf((a / (1.0f + __expf(-a))) * g3[i]);
}

__global__ void init_flags(int* exited, int* exl)
{
  int i = threadIdx.x;
  if (i < B) { exited[i] = 0; exl[i] = -1; }
}

__global__ __launch_bounds__(128)
void gate_pre(const float* __restrict__ x, const float* __restrict__ sw,
              const float* __restrict__ sb, const int* __restrict__ exited,
              int* __restrict__ do_skip, int li)
{
  const int tid = threadIdx.x;
  const int b = tid >> 5, lane = tid & 31;
  const float* xr = x + (size_t)(b * T + T - 1) * D;
  float acc = 0.0f;
  for (int k = lane; k < D; k += 32) acc += xr[k] * sw[k];
  for (int m = 16; m > 0; m >>= 1) acc += __shfl_xor(acc, m, 32);
  __shared__ float sp[B];
  if (lane == 0) sp[b] = 1.0f / (1.0f + __expf(-(acc + sb[0])));
  __syncthreads();
  if (tid == 0) {
    float sum = 0.0f, cnt = 0.0f;
    for (int i = 0; i < B; ++i) if (!exited[i]) { sum += sp[i]; cnt += 1.0f; }
    float mean = sum / fmaxf(cnt, 1.0f);
    do_skip[0] = (li >= MIN_LAYERS && cnt > 0.0f && mean < SKIP_THR) ? 1 : 0;
  }
}

__global__ __launch_bounds__(256)
void logits_last_kernel(const uint16_t* __restrict__ nbf, const float* __restrict__ lm,
                        float* __restrict__ out, const int* __restrict__ gate)
{
  if (gate && gate[0]) return;
  const int b = blockIdx.y;
  const int v = blockIdx.x * 256 + threadIdx.x;
  const uint16_t* xr = nbf + (size_t)(b * T + T - 1) * D;
  __shared__ float sx[D];
  for (int k = threadIdx.x; k < D; k += 256) sx[k] = bf2f(xr[k]);
  __syncthreads();
  float acc = 0.0f;
  #pragma unroll 4
  for (int k = 0; k < D; ++k) acc += sx[k] * lm[(size_t)k * V + v];
  out[(size_t)b * V + v] = acc * (1.0f / TEMP);
}

__global__ __launch_bounds__(256)
void conf_reduce(const float* __restrict__ ll, const float* __restrict__ x,
                 const float* __restrict__ ew, const float* __restrict__ eb,
                 float* __restrict__ conf, float* __restrict__ exitp,
                 const int* __restrict__ gate)
{
  if (gate && gate[0]) return;
  const int b = blockIdx.x, tid = threadIdx.x;
  const float* lr = ll + (size_t)b * V;
  __shared__ float red[256];
  float mx = -1e30f;
  for (int i = tid; i < V; i += 256) mx = fmaxf(mx, lr[i]);
  red[tid] = mx; __syncthreads();
  for (int st = 128; st > 0; st >>= 1) { if (tid < st) red[tid] = fmaxf(red[tid], red[tid + st]); __syncthreads(); }
  mx = red[0]; __syncthreads();
  float s = 0.0f;
  for (int i = tid; i < V; i += 256) s += __expf(lr[i] - mx);
  red[tid] = s; __syncthreads();
  for (int st = 128; st > 0; st >>= 1) { if (tid < st) red[tid] += red[tid + st]; __syncthreads(); }
  float sum = red[0]; __syncthreads();
  const float* xr = x + (size_t)(b * T + T - 1) * D;
  float d = 0.0f;
  for (int i = tid; i < D; i += 256) d += xr[i] * ew[i];
  red[tid] = d; __syncthreads();
  for (int st = 128; st > 0; st >>= 1) { if (tid < st) red[tid] += red[tid + st]; __syncthreads(); }
  if (tid == 0) {
    conf[b] = 1.0f / sum;                              // max of softmax
    exitp[b] = 1.0f / (1.0f + __expf(-(red[0] + eb[0])));
  }
}

__global__ void decide_kernel(const float* __restrict__ conf, const float* __restrict__ exitp,
                              int* __restrict__ exited, int* __restrict__ newly,
                              int* __restrict__ exl, const int* __restrict__ do_skip, int li)
{
  const int b = threadIdx.x;
  if (b >= B) return;
  bool should = (conf[b] > EXIT_THR) || (exitp[b] > EXIT_THR);
  bool nw = should && !exited[b] && !do_skip[0] && (li >= MIN_LAYERS - 1);
  newly[b] = nw ? 1 : 0;
  if (nw) { exited[b] = 1; exl[b] = li; }
}

__global__ void finalize_kernel(const int* __restrict__ exited, int* __restrict__ newly,
                                int* __restrict__ exl, float* __restrict__ out_exl)
{
  const int b = threadIdx.x;
  if (b >= B) return;
  int never = exited[b] ? 0 : 1;
  newly[b] = never;
  int e = never ? (L - 1) : exl[b];
  ((int*)out_exl)[b] = e;                              // int32 output, bit-stored
}

// ---------------------------------------------------------------------------
extern "C" void kernel_launch(void* const* d_in, const int* in_sizes, int n_in,
                              void* d_out, int out_size, void* d_ws, size_t ws_size,
                              hipStream_t stream)
{
  (void)in_sizes; (void)n_in; (void)out_size; (void)ws_size;
  const int*   ids   = (const int*)  d_in[0];
  const float* emb   = (const float*)d_in[1];
  const float* wq    = (const float*)d_in[2];
  const float* wk    = (const float*)d_in[3];
  const float* wv    = (const float*)d_in[4];
  const float* wo    = (const float*)d_in[5];
  const float* w1    = (const float*)d_in[6];
  const float* w2    = (const float*)d_in[7];
  const float* w3    = (const float*)d_in[8];
  const float* n1    = (const float*)d_in[9];
  const float* n2    = (const float*)d_in[10];
  const float* fnorm = (const float*)d_in[11];
  const float* lm    = (const float*)d_in[12];
  const float* ew    = (const float*)d_in[13];
  const float* eb    = (const float*)d_in[14];
  const float* sw    = (const float*)d_in[15];
  const float* sb    = (const float*)d_in[16];

  char* p = (char*)d_ws;
  auto carve = [&](size_t bytes) -> char* {
    char* r = p; p += (bytes + 255) & ~(size_t)255; return r;
  };
  float*    x      = (float*)   carve((size_t)B * T * D * 4);
  float*    tmpf   = (float*)   carve((size_t)B * T * D * 4);
  float*    g1     = (float*)   carve((size_t)B * T * F * 4);
  float*    g3     = (float*)   carve((size_t)B * T * F * 4);
  float*    scores = (float*)   carve((size_t)B * H * T * T * 4);
  uint16_t* hbf    = (uint16_t*)carve((size_t)B * T * D * 2);
  uint16_t* qb     = (uint16_t*)carve((size_t)B * T * D * 2);
  uint16_t* kb     = (uint16_t*)carve((size_t)B * T * D * 2);
  uint16_t* vT     = (uint16_t*)carve((size_t)B * T * D * 2);
  uint16_t* obf    = (uint16_t*)carve((size_t)B * T * D * 2);
  uint16_t* attb   = (uint16_t*)carve((size_t)B * H * T * T * 2);
  uint16_t* gpb    = (uint16_t*)carve((size_t)B * T * F * 2);
  float*    llast  = (float*)   carve((size_t)B * V * 4);
  float*    conf   = (float*)   carve(64);
  float*    exitp  = (float*)   carve(64);
  int*      exited = (int*)     carve(64);
  int*      newly  = (int*)     carve(64);
  int*      exl    = (int*)     carve(64);
  int*      dskip  = (int*)     carve(64);

  float* outL    = (float*)d_out;
  float* out_exl = outL + (size_t)B * T * V;

  init_flags<<<1, 32, 0, stream>>>(exited, exl);
  embed_kernel<<<B * T, 256, 0, stream>>>(ids, emb, x);

  const dim3 gD (D / 256,  MR / 64);     // N=1024 GEMMs
  const dim3 gF (F / 256,  MR / 64);     // N=4096 GEMMs
  const dim3 gV (V / 256,  MR / 64);     // logits GEMM (32000/256 = 125)

  for (int li = 0; li < L; ++li) {
    const float* wq_l = wq + (size_t)li * D * D;
    const float* wk_l = wk + (size_t)li * D * D;
    const float* wv_l = wv + (size_t)li * D * D;
    const float* wo_l = wo + (size_t)li * D * D;
    const float* w1_l = w1 + (size_t)li * D * F;
    const float* w2_l = w2 + (size_t)li * F * D;
    const float* w3_l = w3 + (size_t)li * D * F;

    gate_pre<<<1, 128, 0, stream>>>(x, sw + (size_t)li * D, sb + li, exited, dskip, li);

    // --- attention sub-block (all gated on do_skip) ---
    rmsnorm_kernel<<<B * T, 256, 0, stream>>>(x, n1 + (size_t)li * D, hbf, dskip);
    gemm_bf16<EP_F32><<<gD, 256, 0, stream>>>(hbf, wq_l, MR, D, D, tmpf, nullptr, nullptr, dskip, nullptr);
    rope_kernel<<<(B * T * H * 32) / 256, 256, 0, stream>>>(tmpf, qb, dskip);
    gemm_bf16<EP_F32><<<gD, 256, 0, stream>>>(hbf, wk_l, MR, D, D, tmpf, nullptr, nullptr, dskip, nullptr);
    rope_kernel<<<(B * T * H * 32) / 256, 256, 0, stream>>>(tmpf, kb, dskip);
    gemm_bf16<EP_VT><<<gD, 256, 0, stream>>>(hbf, wv_l, MR, D, D, nullptr, vT, nullptr, dskip, nullptr);

    attn_scores<<<(B * H * (T / 16) * (T / 16)) / 8, 256, 0, stream>>>(qb, kb, scores, dskip);
    softmax_rows<<<B * H * T, 128, 0, stream>>>(scores, attb, dskip);
    attn_pv<<<(B * H * (T / 16) * (HD / 16)) / 8, 256, 0, stream>>>(attb, vT, obf, dskip);
    gemm_bf16<EP_RESID><<<gD, 256, 0, stream>>>(obf, wo_l, MR, D, D, x, nullptr, x, dskip, nullptr);

    // --- FFN sub-block ---
    rmsnorm_kernel<<<B * T, 256, 0, stream>>>(x, n2 + (size_t)li * D, hbf, dskip);
    gemm_bf16<EP_F32><<<gF, 256, 0, stream>>>(hbf, w1_l, MR, F, D, g1, nullptr, nullptr, dskip, nullptr);
    gemm_bf16<EP_F32><<<gF, 256, 0, stream>>>(hbf, w3_l, MR, F, D, g3, nullptr, nullptr, dskip, nullptr);
    silu_mul_kernel<<<(B * T * F) / 256, 256, 0, stream>>>(g1, g3, gpb, dskip, B * T * F);
    gemm_bf16<EP_RESID><<<gD, 256, 0, stream>>>(gpb, w2_l, MR, D, F, x, nullptr, x, dskip, nullptr);

    // --- early-exit logits path ---
    rmsnorm_kernel<<<B * T, 256, 0, stream>>>(x, fnorm, hbf, dskip);
    logits_last_kernel<<<dim3(V / 256, B), 256, 0, stream>>>(hbf, lm, llast, dskip);
    conf_reduce<<<B, 256, 0, stream>>>(llast, x, ew + (size_t)li * D, eb + li, conf, exitp, dskip);
    decide_kernel<<<1, 32, 0, stream>>>(conf, exitp, exited, newly, exl, dskip, li);
    // full logits only for batches that exit at this layer (per-batch predicate)
    gemm_bf16<EP_F32><<<gV, 256, 0, stream>>>(hbf, lm, MR, V, D, outL, nullptr, nullptr, nullptr, newly);
  }

  // final logits for batches that never exited
  rmsnorm_kernel<<<B * T, 256, 0, stream>>>(x, fnorm, hbf, nullptr);
  finalize_kernel<<<1, 32, 0, stream>>>(exited, newly, exl, out_exl);
  gemm_bf16<EP_F32><<<gV, 256, 0, stream>>>(hbf, lm, MR, V, D, outL, nullptr, nullptr, nullptr, newly);
}
